// GroupedQueryAttention_48979807044178
// MI455X (gfx1250) — compile-verified, matrix operand order checked by reading, not run on hardware
//
#include <hip/hip_runtime.h>
#include <hip/hip_bf16.h>
#include <math.h>

// ---------------------------------------------------------------------------
// Types for CDNA5 WMMA (wave32): v16bf A/B fragments, v8f C/D accumulators
// ---------------------------------------------------------------------------
typedef __attribute__((ext_vector_type(16))) __bf16 v16bf;
typedef __attribute__((ext_vector_type(8)))  __bf16 v8bf;
typedef __attribute__((ext_vector_type(4)))  __bf16 v4bf;
typedef __attribute__((ext_vector_type(8)))  float  v8f;

#define D_MODEL 2048
#define NGROUPS 16
#define DHEAD   128
#define BATCH   4
#define SEQ     2048
#define NKEY    2048

__device__ __forceinline__ v8f wmma_bf16(v16bf a, v16bf b, v8f c) {
  // 8 args: (neg_a, A, neg_b, B, c_mod, C, reuse_a, reuse_b)
  return __builtin_amdgcn_wmma_f32_16x16x32_bf16(false, a, false, b, (short)0, c,
                                                 false, false);
}

// Gather a 16x32 bf16 A (or 32x16 B) fragment from LDS laid out row-major
// [row][k] with ldk elements per row. Per the ISA 16-bit layout:
//   lane L: row = L&15,  K = base + {0..7} and base + {16..23}, base = (L>>4)*8
// -> two 16-byte ds loads per lane.
__device__ __forceinline__ v16bf load_frag_lds(const __bf16* base, int row, int ldk) {
  const int lane = threadIdx.x & 31;
  const __bf16* p = base + (size_t)(row + (lane & 15)) * ldk + ((lane >> 4) << 3);
  v8bf lo = *(const v8bf*)p;
  v8bf hi = *(const v8bf*)(p + 16);
  v16bf out;
#pragma unroll
  for (int e = 0; e < 8; ++e) { out[e] = lo[e]; out[e + 8] = hi[e]; }
  return out;
}

// ---------------------------------------------------------------------------
// GEMM: Y = X @ W^T     X:(M,K) f32 or bf16, W:(N,K) f32, Y:(M,N) bf16 or f32
// Block tile 128x128, BK=32, 256 threads = 8 waves, wave tile 64x32 (4x2 WMMA).
// Register double-buffered: next K-chunk's global loads are in flight while
// WMMAs run on the current LDS tile.
// ---------------------------------------------------------------------------
template <bool IN_BF16, bool OUT_BF16>
__global__ __launch_bounds__(256) void proj_kernel(const void* __restrict__ Xv,
                                                   const float* __restrict__ W,
                                                   void* __restrict__ Yv,
                                                   int M, int Nn, int K) {
  __shared__ __bf16 Xs[128][32];  // 8 KB
  __shared__ __bf16 Ws[128][32];  // 8 KB

  const int tid = threadIdx.x;
  const int lane = tid & 31;
  const int wid = tid >> 5;
  const int hlf = lane >> 4;
  const int lr = lane & 15;
  const int m0 = blockIdx.y * 128;
  const int n0 = blockIdx.x * 128;
  const int wm = wid >> 2;  // 0..1  (64 rows each)
  const int wn = wid & 3;   // 0..3  (32 cols each)

  v8f acc[4][2] = {};

  // staging registers for the next K-chunk
  float4 xrf[4];
  v8bf   xrb[2];
  float4 wrf[4];

  auto load_stage = [&](int kt) {
    if (IN_BF16) {
      const __bf16* X = (const __bf16*)Xv;
#pragma unroll
      for (int j = 0; j < 2; ++j) {
        int idx = tid + j * 256;
        int row = idx >> 2, c = idx & 3;
        xrb[j] = *(const v8bf*)(X + (size_t)(m0 + row) * K + (size_t)kt * 32 + c * 8);
      }
    } else {
      const float* X = (const float*)Xv;
#pragma unroll
      for (int j = 0; j < 4; ++j) {
        int idx = tid + j * 256;
        int row = idx >> 3, c = idx & 7;
        xrf[j] = *(const float4*)(X + (size_t)(m0 + row) * K + (size_t)kt * 32 + c * 4);
      }
    }
#pragma unroll
    for (int j = 0; j < 4; ++j) {
      int idx = tid + j * 256;
      int row = idx >> 3, c = idx & 7;
      wrf[j] = *(const float4*)(W + (size_t)(n0 + row) * K + (size_t)kt * 32 + c * 4);
    }
  };

  auto store_stage = [&]() {
    if (IN_BF16) {
#pragma unroll
      for (int j = 0; j < 2; ++j) {
        int idx = tid + j * 256;
        int row = idx >> 2, c = idx & 3;
        *(v8bf*)&Xs[row][c * 8] = xrb[j];
      }
    } else {
#pragma unroll
      for (int j = 0; j < 4; ++j) {
        int idx = tid + j * 256;
        int row = idx >> 3, c = idx & 7;
        v4bf h;
        h[0] = (__bf16)xrf[j].x; h[1] = (__bf16)xrf[j].y;
        h[2] = (__bf16)xrf[j].z; h[3] = (__bf16)xrf[j].w;
        *(v4bf*)&Xs[row][c * 4] = h;
      }
    }
#pragma unroll
    for (int j = 0; j < 4; ++j) {
      int idx = tid + j * 256;
      int row = idx >> 3, c = idx & 7;
      v4bf h;
      h[0] = (__bf16)wrf[j].x; h[1] = (__bf16)wrf[j].y;
      h[2] = (__bf16)wrf[j].z; h[3] = (__bf16)wrf[j].w;
      *(v4bf*)&Ws[row][c * 4] = h;
    }
  };

  const int nk = K >> 5;
  load_stage(0);
  for (int kt = 0; kt < nk; ++kt) {
    store_stage();
    __syncthreads();
    if (kt + 1 < nk) load_stage(kt + 1);  // overlap with WMMA below

    v16bf afrag[4];
#pragma unroll
    for (int i = 0; i < 4; ++i)
      afrag[i] = load_frag_lds(&Xs[0][0], wm * 64 + i * 16, 32);
    v16bf wfrag[2];
#pragma unroll
    for (int j = 0; j < 2; ++j)
      wfrag[j] = load_frag_lds(&Ws[0][0], wn * 32 + j * 16, 32);
#pragma unroll
    for (int i = 0; i < 4; ++i)
#pragma unroll
      for (int j = 0; j < 2; ++j)
        acc[i][j] = wmma_bf16(afrag[i], wfrag[j], acc[i][j]);
    __syncthreads();
  }

  // ---- epilogue: C layout lane=(n) vgpr r=(m&7), m += 8*(lane>>4) ----
#pragma unroll
  for (int i = 0; i < 4; ++i)
#pragma unroll
    for (int j = 0; j < 2; ++j)
#pragma unroll
      for (int r = 0; r < 8; ++r) {
        int m = m0 + wm * 64 + i * 16 + hlf * 8 + r;
        int n = n0 + wn * 32 + j * 16 + lr;
        float v = acc[i][j][r];
        if (OUT_BF16) ((__bf16*)Yv)[(size_t)m * Nn + n] = (__bf16)v;
        else          ((float*)Yv)[(size_t)m * Nn + n] = v;
      }
}

// ---------------------------------------------------------------------------
// RoPE on Q (bf16 in-place): first 512 model dims, half = 256, per the
// reference's pre-head-split rotation.
// ---------------------------------------------------------------------------
__global__ __launch_bounds__(256) void rope_kernel(__bf16* __restrict__ Q,
                                                   const int* __restrict__ pos_ids,
                                                   int S) {
  const int row = blockIdx.x;         // 0 .. B*S-1
  const int s = row % S;
  const int i = threadIdx.x;          // 0..255
  const float pos = (float)pos_ids[s];
  const float inv_freq = __expf(-((2.0f * (float)i) / 512.0f) * 9.210340371976184f);
  const float pe = pos * inv_freq;
  float sn, cs;
  __sincosf(pe, &sn, &cs);
  __bf16* p = Q + (size_t)row * D_MODEL;
  const float a = (float)p[i];
  const float b = (float)p[i + 256];
  p[i]       = (__bf16)(a * cs - b * sn);
  p[i + 256] = (__bf16)(b * cs + a * sn);
}

// ---------------------------------------------------------------------------
// Flash attention: one block = (b,g) x 128 query rows; 8 waves x 16 rows.
// Streams 32-key chunks; 16 WMMAs / wave / chunk (8 for QK^T, 8 for PV).
// Register double-buffered K/V staging; all fragments hoisted so each WMMA
// burst issues against a single ds wait.
// ---------------------------------------------------------------------------
__global__ __launch_bounds__(256) void attn_kernel(const __bf16* __restrict__ Qbf,
                                                   const __bf16* __restrict__ Kbf,
                                                   const __bf16* __restrict__ Vbf,
                                                   const int* __restrict__ mask,
                                                   __bf16* __restrict__ Obf) {
  __shared__ __bf16 Ks[32][128];      // [key][dh]  8 KB  (B operand for QK^T)
  __shared__ __bf16 Vts[128][32];     // [dh][key]  8 KB  (B operand for PV)
  __shared__ __bf16 Ps[8][16][32];    // per-wave P staging, 8 KB

  const int tid = threadIdx.x;
  const int lane = tid & 31;
  const int wid = tid >> 5;
  const int hlf = lane >> 4;
  const int lr = lane & 15;
  const int bg = blockIdx.x;
  const int b = bg / NGROUPS;
  const int g = bg % NGROUPS;
  const int q0 = blockIdx.y * 128;
  const float scale = 0.08838834764831845f;  // 1/sqrt(128)
  const float NEG_INF = -__builtin_inff();

  // ---- preload this wave's 16 query rows as 4 A fragments (dh chunks) ----
  v16bf qf[4];
  {
    const __bf16* qbase = Qbf + (size_t)(b * SEQ + q0 + wid * 16 + lr) * D_MODEL +
                          g * DHEAD + (hlf << 3);
#pragma unroll
    for (int c = 0; c < 4; ++c) {
      v8bf lo = *(const v8bf*)(qbase + c * 32);
      v8bf hi = *(const v8bf*)(qbase + c * 32 + 16);
#pragma unroll
      for (int e = 0; e < 8; ++e) { qf[c][e] = lo[e]; qf[c][e + 8] = hi[e]; }
    }
  }

  v8f o[8] = {};          // 16 q rows x 128 dh accumulator (C layout per tile)
  float rmax[8], rsum[8];
#pragma unroll
  for (int r = 0; r < 8; ++r) { rmax[r] = NEG_INF; rsum[r] = 0.0f; }

  // K/V staging registers (next chunk in flight during compute)
  v8bf kreg[2], vreg[2];
  auto load_kv = [&](int kc) {
#pragma unroll
    for (int j = 0; j < 2; ++j) {
      int idx = tid + j * 256;
      int row = idx >> 4;           // key within chunk
      int seg = idx & 15;           // 8 dh cols each
      size_t goff = (size_t)(b * NKEY + kc * 32 + row) * D_MODEL + g * DHEAD + seg * 8;
      kreg[j] = *(const v8bf*)(Kbf + goff);
      vreg[j] = *(const v8bf*)(Vbf + goff);
    }
  };
  auto store_kv = [&]() {
#pragma unroll
    for (int j = 0; j < 2; ++j) {
      int idx = tid + j * 256;
      int row = idx >> 4;
      int seg = idx & 15;
      *(v8bf*)&Ks[row][seg * 8] = kreg[j];
#pragma unroll
      for (int e = 0; e < 8; ++e) Vts[seg * 8 + e][row] = vreg[j][e];
    }
  };

  const int nchunks = NKEY / 32;
  load_kv(0);
  for (int kc = 0; kc < nchunks; ++kc) {
    store_kv();
    __syncthreads();
    if (kc + 1 < nchunks) load_kv(kc + 1);  // overlap with WMMA below

    // ---- scores: all 8 K fragments hoisted, then 8-WMMA burst ----
    v16bf kb[8];
#pragma unroll
    for (int c = 0; c < 4; ++c) {
      kb[c * 2]     = load_frag_lds(&Ks[0][0] + c * 32, 0, 128);
      kb[c * 2 + 1] = load_frag_lds(&Ks[0][0] + c * 32, 16, 128);
    }
    v8f sc[2] = {};
#pragma unroll
    for (int c = 0; c < 4; ++c) {
      sc[0] = wmma_bf16(qf[c], kb[c * 2],     sc[0]);
      sc[1] = wmma_bf16(qf[c], kb[c * 2 + 1], sc[1]);
    }

    // ---- online softmax (rows live within one 16-lane half) ----
    const int mk0 = mask[b * NKEY + kc * 32 + lr];
    const int mk1 = mask[b * NKEY + kc * 32 + 16 + lr];
    float corr[8];
#pragma unroll
    for (int r = 0; r < 8; ++r) {
      float v0 = mk0 ? sc[0][r] * scale : NEG_INF;
      float v1 = mk1 ? sc[1][r] * scale : NEG_INF;
      float mx = fmaxf(v0, v1);
#pragma unroll
      for (int d = 1; d < 16; d <<= 1) mx = fmaxf(mx, __shfl_xor(mx, d, 32));
      float nm = fmaxf(rmax[r], mx);
      float nms = (nm == NEG_INF) ? 0.0f : nm;
      float cr = __expf(rmax[r] - nms);
      float p0 = __expf(v0 - nms);
      float p1 = __expf(v1 - nms);
      Ps[wid][hlf * 8 + r][lr]      = (__bf16)p0;
      Ps[wid][hlf * 8 + r][16 + lr] = (__bf16)p1;
      float ps = p0 + p1;
#pragma unroll
      for (int d = 1; d < 16; d <<= 1) ps += __shfl_xor(ps, d, 32);
      rsum[r] = rsum[r] * cr + ps;
      rmax[r] = nm;
      corr[r] = cr;
    }
#pragma unroll
    for (int t = 0; t < 8; ++t)
#pragma unroll
      for (int r = 0; r < 8; ++r) o[t][r] *= corr[r];

    // ---- PV: hoist all 8 V^T fragments, then wait once for P stores ----
    v16bf vf[8];
#pragma unroll
    for (int t = 0; t < 8; ++t)
      vf[t] = load_frag_lds(&Vts[0][0], t * 16, 32);

    // wave-local LDS RAW: P stores above -> A-fragment reload below
    asm volatile("s_wait_dscnt 0" ::: "memory");
    v16bf pf = load_frag_lds(&Ps[wid][0][0], 0, 32);
#pragma unroll
    for (int t = 0; t < 8; ++t)
      o[t] = wmma_bf16(pf, vf[t], o[t]);
    __syncthreads();
  }

  // ---- epilogue: normalize and store bf16 (B,S,D) with head g at g*128 ----
#pragma unroll
  for (int r = 0; r < 8; ++r) {
    float inv = 1.0f / rsum[r];
    size_t rowoff = (size_t)(b * SEQ + q0 + wid * 16 + hlf * 8 + r) * D_MODEL +
                    g * DHEAD + lr;
#pragma unroll
    for (int t = 0; t < 8; ++t)
      Obf[rowoff + t * 16] = (__bf16)(o[t][r] * inv);
  }
}

// ---------------------------------------------------------------------------
// Launch
// ---------------------------------------------------------------------------
extern "C" void kernel_launch(void* const* d_in, const int* in_sizes, int n_in,
                              void* d_out, int out_size, void* d_ws, size_t ws_size,
                              hipStream_t stream) {
  (void)in_sizes; (void)n_in; (void)out_size; (void)ws_size;
  const float* query = (const float*)d_in[0];
  const float* key   = (const float*)d_in[1];
  const float* value = (const float*)d_in[2];
  const int*   mask  = (const int*)d_in[3];
  const int*   pos   = (const int*)d_in[4];
  const float* Wq    = (const float*)d_in[5];
  const float* Wk    = (const float*)d_in[6];
  const float* Wv    = (const float*)d_in[7];
  const float* Wo    = (const float*)d_in[8];
  float* out = (float*)d_out;

  const int M = BATCH * SEQ;  // 8192
  __bf16* Qbf = (__bf16*)d_ws;
  __bf16* Kbf = Qbf + (size_t)M * D_MODEL;
  __bf16* Vbf = Kbf + (size_t)M * D_MODEL;
  __bf16* Abf = Vbf + (size_t)M * D_MODEL;

  dim3 pgrid(D_MODEL / 128, M / 128);      // (16, 64)
  proj_kernel<false, true><<<pgrid, 256, 0, stream>>>(query, Wq, Qbf, M, D_MODEL, D_MODEL);
  proj_kernel<false, true><<<pgrid, 256, 0, stream>>>(key,   Wk, Kbf, M, D_MODEL, D_MODEL);
  proj_kernel<false, true><<<pgrid, 256, 0, stream>>>(value, Wv, Vbf, M, D_MODEL, D_MODEL);

  rope_kernel<<<M, 256, 0, stream>>>(Qbf, pos, SEQ);

  attn_kernel<<<dim3(BATCH * NGROUPS, SEQ / 128), 256, 0, stream>>>(Qbf, Kbf, Vbf, mask, Abf);

  proj_kernel<true, false><<<pgrid, 256, 0, stream>>>(Abf, Wo, out, M, D_MODEL, D_MODEL);
}
